// Decoder_35424890258098
// MI455X (gfx1250) — compile-verified, matrix operand order checked by reading, not run on hardware
//
#include <hip/hip_runtime.h>
#include <hip/hip_bf16.h>
#include <stdint.h>

// Problem constants (match reference)
#define Vv   32000
#define Ee   512
#define Hh   512
#define Bb   64
#define Ss   64
#define G4H  2048   // 4*H

typedef __attribute__((ext_vector_type(16))) __bf16 v16bf;
typedef __attribute__((ext_vector_type(8)))  float  v8f;

union FragAB { v16bf v; uint4 q[2]; };
union FragC  { v8f   v; float f[8]; };

__device__ __forceinline__ uint16_t f32_to_bf16(float x) {
  uint32_t u = __builtin_bit_cast(uint32_t, x);
  u += 0x7FFFu + ((u >> 16) & 1u);           // round-to-nearest-even
  return (uint16_t)(u >> 16);
}

// Load a 16x32 bf16 fragment (rows striped lane 0..15; half-lane K split:
// lanes 0-15 hold K {0..7,16..23}, lanes 16-31 hold K {8..15,24..31})
// Two 16B loads per lane. Works for A (rows of X/h) and for B (rows of W,
// since we compute A @ W^T and W rows are the B-matrix columns).
__device__ __forceinline__ void load_frag(const uint16_t* __restrict__ base,
                                          int ld, int k, FragAB& f) {
  const int lane = threadIdx.x & 31;
  const int r    = lane & 15;
  const int hi   = lane >> 4;
  const uint16_t* p = base + (size_t)r * ld + k + hi * 8;
  f.q[0] = *(const uint4*)p;
  f.q[1] = *(const uint4*)(p + 16);
}

// ---------------------------------------------------------------------------
// Generic batched GEMM: Cout = (Cin?) + A(M,K)bf16 @ W(N,K)bf16^T  -> f32
// One wave computes a 16(M) x 64(N) tile (4 accumulators, A reused 4x).
// ---------------------------------------------------------------------------
__global__ __launch_bounds__(256)
void gemm_batched_wmma(const uint16_t* __restrict__ A, long long sA, int lda,
                       const uint16_t* __restrict__ W, long long sW, int ldw,
                       const float* __restrict__ Cin, long long sCin,
                       float* __restrict__ Cout, long long sCout,
                       int M, int N, int K, int ldc, int nbatch)
{
  const int gwave = (int)(((size_t)blockIdx.x * blockDim.x + threadIdx.x) >> 5);
  const int lane  = threadIdx.x & 31;
  const int mt = M >> 4, nt = N >> 6;
  const int per = mt * nt;
  const int batch = gwave / per;
  if (batch >= nbatch) return;                 // wave-uniform: EXEC stays full
  const int tile  = gwave - batch * per;
  const int mtile = tile / nt;
  const int ntile = tile - mtile * nt;

  const uint16_t* Ab = A + (size_t)batch * sA + (size_t)(mtile * 16) * lda;
  const uint16_t* Wb = W + (size_t)batch * sW + (size_t)(ntile * 64) * ldw;

  const int c = lane & 15, hi = lane >> 4;

  FragC acc[4];
  if (Cin) {
    const float* cp = Cin + (size_t)batch * sCin
                    + (size_t)(mtile * 16 + hi * 8) * ldc + ntile * 64 + c;
#pragma unroll
    for (int sub = 0; sub < 4; ++sub)
#pragma unroll
      for (int i = 0; i < 8; ++i)
        acc[sub].f[i] = cp[(size_t)i * ldc + sub * 16];
  } else {
#pragma unroll
    for (int sub = 0; sub < 4; ++sub)
#pragma unroll
      for (int i = 0; i < 8; ++i)
        acc[sub].f[i] = 0.0f;
  }

  for (int k = 0; k < K; k += 32) {
    FragAB a; load_frag(Ab, lda, k, a);
#pragma unroll
    for (int sub = 0; sub < 4; ++sub) {
      FragAB b; load_frag(Wb + (size_t)(sub * 16) * ldw, ldw, k, b);
      acc[sub].v = __builtin_amdgcn_wmma_f32_16x16x32_bf16(
          false, a.v, false, b.v, (short)0, acc[sub].v, false, false);
    }
  }

  float* op = Cout + (size_t)batch * sCout
            + (size_t)(mtile * 16 + hi * 8) * ldc + ntile * 64 + c;
#pragma unroll
  for (int sub = 0; sub < 4; ++sub)
#pragma unroll
    for (int i = 0; i < 8; ++i)
      op[(size_t)i * ldc + sub * 16] = acc[sub].f[i];
}

// ---------------------------------------------------------------------------
// Final FC: out[(b*S+s)*V + n] = outs[s,b,:] . fc_w[n,:] + fc_b[n]
// A (4096,1024)bf16 staged through LDS (16xKT tile shared by 8 waves),
// W (32000,1024)bf16 streamed from L2. Permute fused into the store.
// ---------------------------------------------------------------------------
#define FC_KT 64
__global__ __launch_bounds__(256)
void fc_wmma(const uint16_t* __restrict__ A, const uint16_t* __restrict__ W,
             const float* __restrict__ bias, float* __restrict__ out)
{
  __shared__ __align__(16) uint16_t As[16 * FC_KT];       // 2 KB
  const int nblocks = (Vv + 511) / 512;                   // 63
  const int mb = blockIdx.x / nblocks;
  const int nb = blockIdx.x % nblocks;
  const int wave = threadIdx.x >> 5;
  const int lane = threadIdx.x & 31;
  const int n_base = nb * 512 + wave * 64;
  const bool active = (n_base < Vv);                      // wave-uniform
  const uint16_t* Ab = A + (size_t)(mb * 16) * 1024;
  const uint16_t* Wb = W + (size_t)n_base * 1024;

  FragC acc[4];
#pragma unroll
  for (int sub = 0; sub < 4; ++sub)
#pragma unroll
    for (int i = 0; i < 8; ++i) acc[sub].f[i] = 0.0f;

  for (int k0 = 0; k0 < 1024; k0 += FC_KT) {
    {   // cooperative stage: 16x64 bf16 = 2 KB, 8 bytes per thread
      const int t = threadIdx.x;
      const int r = t >> 4;
      const int kk = (t & 15) << 2;
      *(uint2*)(As + r * FC_KT + kk) =
          *(const uint2*)(Ab + (size_t)r * 1024 + k0 + kk);
    }
    __syncthreads();
    if (active) {
#pragma unroll
      for (int k = 0; k < FC_KT; k += 32) {
        FragAB a; load_frag(As, FC_KT, k, a);             // ds_load_b128
#pragma unroll
        for (int sub = 0; sub < 4; ++sub) {
          FragAB b; load_frag(Wb + (size_t)(sub * 16) * 1024, 1024, k0 + k, b);
          acc[sub].v = __builtin_amdgcn_wmma_f32_16x16x32_bf16(
              false, a.v, false, b.v, (short)0, acc[sub].v, false, false);
        }
      }
      if (k0 + FC_KT < 1024)
        __builtin_prefetch(Wb + k0 + FC_KT, 0, 1);        // global_prefetch
    }
    __syncthreads();
  }

  if (active) {
    const int c = lane & 15, hi = lane >> 4;
#pragma unroll
    for (int sub = 0; sub < 4; ++sub) {
      const int col = n_base + sub * 16 + c;
      const float bb = bias[col];
#pragma unroll
      for (int i = 0; i < 8; ++i) {
        const int m = mb * 16 + hi * 8 + i;               // m = s*B + b
        const int s = m >> 6;
        const int b = m & 63;
        out[((size_t)(b * Ss + s)) * Vv + col] = acc[sub].f[i] + bb;
      }
    }
  }
}

// ---------------------------------------------------------------------------
// Elementwise gate activation + state update for one layer (both directions)
// gates layout (2, B, 4H) with PyTorch gate order i,f,g,o along 4H.
// ---------------------------------------------------------------------------
__global__ __launch_bounds__(256)
void lstm_update(const float* __restrict__ gbuf, const float* __restrict__ bsum,
                 float* __restrict__ cst, uint16_t* __restrict__ h_bf,
                 uint16_t* __restrict__ xcat, int cell_base)
{
  const int idx = blockIdx.x * 256 + threadIdx.x;         // < 2*B*H
  const int j   = idx & (Hh - 1);
  const int b   = (idx >> 9) & (Bb - 1);
  const int dir = idx >> 15;
  const float* g  = gbuf + ((size_t)dir * Bb + b) * G4H;
  const float* bs = bsum + (size_t)dir * G4H;
  const float gi = g[j]          + bs[j];
  const float gf = g[j + Hh]     + bs[j + Hh];
  const float gg = g[j + 2 * Hh] + bs[j + 2 * Hh];
  const float go = g[j + 3 * Hh] + bs[j + 3 * Hh];
  const float si = 1.0f / (1.0f + __expf(-gi));
  const float sf = 1.0f / (1.0f + __expf(-gf));
  const float so = 1.0f / (1.0f + __expf(-go));
  const int cell = cell_base + dir;
  const size_t ci = ((size_t)cell * Bb + b) * Hh + j;
  const float cn = sf * cst[ci] + si * tanhf(gg);
  cst[ci] = cn;
  const float hn = so * tanhf(cn);
  const uint16_t hb = f32_to_bf16(hn);
  h_bf[ci] = hb;
  xcat[(size_t)b * (2 * Hh) + dir * Hh + j] = hb;          // concat [fwd,bwd]
}

// Embedding gather with teacher-forcing index: idx(s=0)=Y[b][1], else Y[b][s-1]
__global__ __launch_bounds__(256)
void gather_emb(const int* __restrict__ Y, const float* __restrict__ emb,
                uint16_t* __restrict__ xbf)
{
  const size_t i = (size_t)blockIdx.x * 256 + threadIdx.x; // grid == S*B*E
  const int e = (int)(i & (Ee - 1));
  const int b = (int)((i >> 9) & (Bb - 1));
  const int s = (int)(i >> 15);
  const int idx = (s == 0) ? Y[b * Ss + 1] : Y[b * Ss + s - 1];
  xbf[i] = f32_to_bf16(emb[(size_t)idx * Ee + e]);
}

__global__ __launch_bounds__(256)
void cvt_bf16(const float* __restrict__ src, uint16_t* __restrict__ dst,
              long long n)
{
  for (long long i = (long long)blockIdx.x * 256 + threadIdx.x; i < n;
       i += (long long)gridDim.x * 256)
    dst[i] = f32_to_bf16(src[i]);
}

__global__ __launch_bounds__(256)
void addvec(const float* __restrict__ a, const float* __restrict__ b,
            float* __restrict__ o, int n)
{
  const int i = blockIdx.x * 256 + threadIdx.x;
  if (i < n) o[i] = a[i] + b[i];
}

__global__ __launch_bounds__(256)
void copyf(const float* __restrict__ a, float* __restrict__ o, int n)
{
  const int i = blockIdx.x * 256 + threadIdx.x;
  if (i < n) o[i] = a[i];
}

// ---------------------------------------------------------------------------
extern "C" void kernel_launch(void* const* d_in, const int* in_sizes, int n_in,
                              void* d_out, int out_size, void* d_ws, size_t ws_size,
                              hipStream_t stream)
{
  const float* hidden = (const float*)d_in[0];
  const float* cell   = (const float*)d_in[1];
  const int*   Y      = (const int*)d_in[2];
  const float* emb    = (const float*)d_in[3];
  const float* wih0   = (const float*)d_in[4];
  const float* whh0   = (const float*)d_in[5];
  const float* bih0   = (const float*)d_in[6];
  const float* bhh0   = (const float*)d_in[7];
  const float* wih1   = (const float*)d_in[8];
  const float* whh1   = (const float*)d_in[9];
  const float* bih1   = (const float*)d_in[10];
  const float* bhh1   = (const float*)d_in[11];
  const float* fcw    = (const float*)d_in[12];
  const float* fcb    = (const float*)d_in[13];
  float* out = (float*)d_out;

  char* ws = (char*)d_ws;
  size_t off = 0;
  auto carve = [&](size_t bytes) -> void* {
    void* p = ws + off;
    off = (off + bytes + 255) & ~(size_t)255;
    return p;
  };
  uint16_t* wih0_bf = (uint16_t*)carve((size_t)2 * G4H * Ee  * 2);
  uint16_t* whh0_bf = (uint16_t*)carve((size_t)2 * G4H * Hh  * 2);
  uint16_t* wih1_bf = (uint16_t*)carve((size_t)2 * G4H * 2 * Hh * 2);
  uint16_t* whh1_bf = (uint16_t*)carve((size_t)2 * G4H * Hh  * 2);
  uint16_t* fcw_bf  = (uint16_t*)carve((size_t)Vv * 2 * Hh * 2);
  uint16_t* xbf     = (uint16_t*)carve((size_t)Ss * Bb * Ee * 2);
  float*    g0pre   = (float*)   carve((size_t)2 * Ss * Bb * G4H * 4);
  float*    b0      = (float*)   carve((size_t)2 * G4H * 4);
  float*    b1      = (float*)   carve((size_t)2 * G4H * 4);
  uint16_t* h_bf    = (uint16_t*)carve((size_t)4 * Bb * Hh * 2);
  float*    cst     = (float*)   carve((size_t)4 * Bb * Hh * 4);
  float*    gbuf    = (float*)   carve((size_t)2 * Bb * G4H * 4);
  uint16_t* x1bf    = (uint16_t*)carve((size_t)Bb * 2 * Hh * 2);
  uint16_t* outsbf  = (uint16_t*)carve((size_t)Ss * Bb * 2 * Hh * 2);

  const int T = 256;

  // One-time (per call) conversions / setup
  cvt_bf16<<<2048, T, 0, stream>>>(wih0, wih0_bf, (long long)2 * G4H * Ee);
  cvt_bf16<<<2048, T, 0, stream>>>(whh0, whh0_bf, (long long)2 * G4H * Hh);
  cvt_bf16<<<2048, T, 0, stream>>>(wih1, wih1_bf, (long long)2 * G4H * 2 * Hh);
  cvt_bf16<<<2048, T, 0, stream>>>(whh1, whh1_bf, (long long)2 * G4H * Hh);
  cvt_bf16<<<4096, T, 0, stream>>>(fcw,  fcw_bf,  (long long)Vv * 2 * Hh);
  cvt_bf16<<<512,  T, 0, stream>>>(hidden, h_bf,  (long long)4 * Bb * Hh);
  copyf  <<<512,  T, 0, stream>>>(cell, cst, 4 * Bb * Hh);
  addvec <<<16,   T, 0, stream>>>(bih0, bhh0, b0, 2 * G4H);
  addvec <<<16,   T, 0, stream>>>(bih1, bhh1, b1, 2 * G4H);
  gather_emb<<<(Ss * Bb * Ee) / T, T, 0, stream>>>(Y, emb, xbf);

  // Precompute layer-0 input gates for ALL steps, both dirs (parallel GEMM):
  // g0pre[dir] (S*B, 4H) = X (S*B, E) @ wih0[dir]^T
  {
    const int Mv = Ss * Bb, Nv = G4H, Kv = Ee;
    const int waves = 2 * (Mv / 16) * (Nv / 64);
    gemm_batched_wmma<<<(waves + 7) / 8, T, 0, stream>>>(
        xbf, 0, Ee,
        wih0_bf, (long long)G4H * Ee, Ee,
        nullptr, 0,
        g0pre, (long long)Mv * Nv,
        Mv, Nv, Kv, Nv, 2);
  }

  // Sequential recurrence: 64 steps, all launches graph-captured
  for (int t = 0; t < Ss; ++t) {
    // Layer 0: gates = g0pre[t] + h0 @ whh0^T   (both dirs batched)
    {
      const int waves = 2 * (Bb / 16) * (G4H / 64);
      gemm_batched_wmma<<<(waves + 7) / 8, T, 0, stream>>>(
          h_bf, (long long)Bb * Hh, Hh,
          whh0_bf, (long long)G4H * Hh, Hh,
          g0pre + (size_t)t * Bb * G4H, (long long)(Ss * Bb) * G4H,
          gbuf, (long long)Bb * G4H,
          Bb, G4H, Hh, G4H, 2);
    }
    lstm_update<<<(2 * Bb * Hh) / T, T, 0, stream>>>(gbuf, b0, cst, h_bf, x1bf, 0);

    // Layer 1a: gates = x1 @ wih1^T
    {
      const int waves = 2 * (Bb / 16) * (G4H / 64);
      gemm_batched_wmma<<<(waves + 7) / 8, T, 0, stream>>>(
          x1bf, 0, 2 * Hh,
          wih1_bf, (long long)G4H * 2 * Hh, 2 * Hh,
          nullptr, 0,
          gbuf, (long long)Bb * G4H,
          Bb, G4H, 2 * Hh, G4H, 2);
    }
    // Layer 1b: gates += h1 @ whh1^T
    {
      const int waves = 2 * (Bb / 16) * (G4H / 64);
      gemm_batched_wmma<<<(waves + 7) / 8, T, 0, stream>>>(
          h_bf + (size_t)2 * Bb * Hh, (long long)Bb * Hh, Hh,
          whh1_bf, (long long)G4H * Hh, Hh,
          gbuf, (long long)Bb * G4H,
          gbuf, (long long)Bb * G4H,
          Bb, G4H, Hh, G4H, 2);
    }
    lstm_update<<<(2 * Bb * Hh) / T, T, 0, stream>>>(
        gbuf, b1, cst, h_bf, outsbf + (size_t)t * Bb * 2 * Hh, 2);
  }

  // Final FC with fused (S,B,V)->(B,S,V) permute + bias
  {
    const int nblocks = (Vv + 511) / 512;            // 63
    const int grid = (Ss * Bb / 16) * nblocks;       // 256 * 63
    fc_wmma<<<grid, T, 0, stream>>>(outsbf, fcw_bf, fcb, out);
  }
}
